// CausalTransformerBlock_67010079752363
// MI455X (gfx1250) — matrix operand order repaired and checked by reading, not compiled, not run on hardware
//
#include <hip/hip_runtime.h>
#include <hip/hip_bf16.h>

typedef _Float16 v8h  __attribute__((ext_vector_type(8)));
typedef _Float16 v16h __attribute__((ext_vector_type(16)));
typedef float    v8f  __attribute__((ext_vector_type(8)));

#define DIM_C   1024
#define HEADS_C 16
#define HD_C    64
#define FFN_C   4096
#define B_C     2
#define T_C     2048
#define M_C     4096      // B*T
#define WIN_C   125

// ---------------------------------------------------------------- f32 -> f16
__global__ __launch_bounds__(256)
void xb_f32_to_f16(const float* __restrict__ in, _Float16* __restrict__ out, int n) {
    int i = blockIdx.x * 256 + threadIdx.x;
    if (i < n) out[i] = (_Float16)in[i];
}

// ---------------------------------------------------------------- layernorm -> f16
__global__ __launch_bounds__(256)
void xb_layernorm_f16(const float* __restrict__ x, const float* __restrict__ w,
                      const float* __restrict__ b, _Float16* __restrict__ out) {
    __shared__ float red0[8];
    __shared__ float red1[8];
    int row  = blockIdx.x;
    int tid  = threadIdx.x;
    const float* xr = x + (size_t)row * DIM_C;
    float v[4];
    float s = 0.f, ss = 0.f;
#pragma unroll
    for (int i = 0; i < 4; ++i) {
        v[i] = xr[tid + i * 256];
        s  += v[i];
        ss += v[i] * v[i];
    }
#pragma unroll
    for (int off = 16; off > 0; off >>= 1) {
        s  += __shfl_xor(s,  off);
        ss += __shfl_xor(ss, off);
    }
    int wid = tid >> 5, lane = tid & 31;
    if (lane == 0) { red0[wid] = s; red1[wid] = ss; }
    __syncthreads();
    float ts = 0.f, tss = 0.f;
#pragma unroll
    for (int i = 0; i < 8; ++i) { ts += red0[i]; tss += red1[i]; }
    float mean = ts * (1.f / DIM_C);
    float var  = tss * (1.f / DIM_C) - mean * mean;
    float rstd = rsqrtf(var + 1e-5f);
#pragma unroll
    for (int i = 0; i < 4; ++i) {
        int c = tid + i * 256;
        float y = (v[i] - mean) * rstd * w[c] + b[c];
        out[(size_t)row * DIM_C + c] = (_Float16)y;
    }
}

// ---------------------------------------------------------------- WMMA GEMM
// Y[M,N] = A[M,K](f16) @ W[N,K]^T(f16), f32 accumulate.
// MODE 0: f32 out, no bias. MODE 1: f32 out + bias. MODE 2: f16 out, bias + exact GELU.
// Wave tile 32x64 (2x4 accumulators), 8 waves/block -> 128x128 block tile.
// k-loop: 64-wide outer step (16 WMMAs/iter) + global_prefetch at distance 2 iters.
template <int MODE>
__global__ __launch_bounds__(256)
void xb_gemm_wmma(const _Float16* __restrict__ A, const _Float16* __restrict__ W,
                  const float* __restrict__ bias, void* __restrict__ outv,
                  int M, int N, int K) {
    int lane = threadIdx.x & 31;
    int wid  = threadIdx.x >> 5;
    int l16  = lane & 15;
    int lh   = lane >> 4;
    int m_base = blockIdx.y * 128 + (wid & 3) * 32;
    int n_base = blockIdx.x * 128 + (wid >> 2) * 64;

    const _Float16* a0 = A + (size_t)(m_base + l16) * K;
    const _Float16* a1 = a0 + (size_t)16 * K;
    const _Float16* b0 = W + (size_t)(n_base + l16) * K;

    v8f acc[2][4] = {};

    for (int kb = 0; kb < K; kb += 64) {
        // prefetch two outer iterations ahead (256 B per A/B row stream)
        if (kb + 128 < K) {
            __builtin_prefetch(a0 + kb + 128, 0, 1);
            __builtin_prefetch(a1 + kb + 128, 0, 1);
#pragma unroll
            for (int j = 0; j < 4; ++j)
                __builtin_prefetch(b0 + (size_t)(j * 16) * K + kb + 128, 0, 1);
        }
#pragma unroll
        for (int ku = 0; ku < 2; ++ku) {
            int k0 = kb + ku * 32;
            v16h af[2];
            {
                const _Float16* p = a0 + k0 + lh * 8;
                v8h lo = *(const v8h*)(p);
                v8h hi = *(const v8h*)(p + 16);
                af[0] = __builtin_shufflevector(lo, hi, 0,1,2,3,4,5,6,7,8,9,10,11,12,13,14,15);
                const _Float16* q = a1 + k0 + lh * 8;
                v8h lo1 = *(const v8h*)(q);
                v8h hi1 = *(const v8h*)(q + 16);
                af[1] = __builtin_shufflevector(lo1, hi1, 0,1,2,3,4,5,6,7,8,9,10,11,12,13,14,15);
            }
            v16h bf[4];
#pragma unroll
            for (int j = 0; j < 4; ++j) {
                const _Float16* p = b0 + (size_t)(j * 16) * K + k0 + lh * 8;
                v8h lo = *(const v8h*)(p);
                v8h hi = *(const v8h*)(p + 16);
                bf[j] = __builtin_shufflevector(lo, hi, 0,1,2,3,4,5,6,7,8,9,10,11,12,13,14,15);
            }
#pragma unroll
            for (int i = 0; i < 2; ++i)
#pragma unroll
                for (int j = 0; j < 4; ++j)
                    acc[i][j] = __builtin_amdgcn_wmma_f32_16x16x32_f16(
                        false, af[i], false, bf[j], (short)0, acc[i][j], false, false);
        }
    }

#pragma unroll
    for (int i = 0; i < 2; ++i) {
#pragma unroll
        for (int j = 0; j < 4; ++j) {
            int n = n_base + j * 16 + l16;
            float bv = 0.f;
            if (MODE != 0) bv = bias[n];
#pragma unroll
            for (int v = 0; v < 8; ++v) {
                int m = m_base + i * 16 + lh * 8 + v;
                float val = acc[i][j][v];
                if (MODE != 0) val += bv;     // no +0.0 VALU ops in MODE 0
                if (MODE == 2) {
                    val = 0.5f * val * (1.f + erff(val * 0.70710678118654752f));
                    ((_Float16*)outv)[(size_t)m * N + n] = (_Float16)val;
                } else {
                    ((float*)outv)[(size_t)m * N + n] = val;
                }
            }
        }
    }
}

// ---------------------------------------------------------------- RoPE (in-place on Q and K)
__global__ __launch_bounds__(256)
void xb_rope(float* __restrict__ Q, float* __restrict__ K) {
    int idx  = blockIdx.x * 256 + threadIdx.x;   // B*T*H*32 threads
    int i    = idx & 31;                          // pair index within head
    int rest = idx >> 5;
    int h    = rest & (HEADS_C - 1);
    int t    = (rest >> 4) & (T_C - 1);
    int b    = rest >> 15;
    size_t base = ((size_t)(b * T_C + t)) * DIM_C + h * HD_C;
    // inv_freq = 10000^(-i/32) = exp(-i * ln(10000)/32)
    float ang = (float)t * __expf(-(float)i * (9.21034037197618f / 32.f));
    float s, c;
    __sincosf(ang, &s, &c);
    float q1 = Q[base + i], q2 = Q[base + i + 32];
    Q[base + i]      = q1 * c - q2 * s;
    Q[base + i + 32] = q2 * c + q1 * s;
    float k1 = K[base + i], k2 = K[base + i + 32];
    K[base + i]      = k1 * c - k2 * s;
    K[base + i + 32] = k2 * c + k1 * s;
}

// ---------------------------------------------------------------- sliding-window attention
// One wave32 per (b,h,t) query row; f32 softmax; output f16 [B,T,H,hd].
__global__ __launch_bounds__(128)
void xb_attn(const float* __restrict__ Q, const float* __restrict__ K,
             const float* __restrict__ V, _Float16* __restrict__ O) {
    __shared__ float qsh[4][64];
    int lane = threadIdx.x & 31;
    int widx = threadIdx.x >> 5;
    int gw = blockIdx.x * 4 + widx;
    int t = gw & (T_C - 1);
    int h = (gw >> 11) & (HEADS_C - 1);
    int b = gw >> 15;
    size_t rowq = ((size_t)(b * T_C + t)) * DIM_C + h * HD_C;

    qsh[widx][lane]      = Q[rowq + lane];
    qsh[widx][lane + 32] = Q[rowq + lane + 32];
    __syncthreads();

    int s0 = t - (WIN_C - 1); if (s0 < 0) s0 = 0;
    int cnt = t - s0 + 1;                 // <= 125

    float sc[4];
#pragma unroll
    for (int j = 0; j < 4; ++j) {
        int idx = j * 32 + lane;
        if (idx < cnt) {
            const float* kp = K + ((size_t)(b * T_C + s0 + idx)) * DIM_C + h * HD_C;
            float acc = 0.f;
#pragma unroll 16
            for (int d = 0; d < 64; ++d) acc += qsh[widx][d] * kp[d];
            sc[j] = acc * 0.125f;         // 1/sqrt(64)
        } else {
            sc[j] = -1e30f;
        }
    }
    float mx = fmaxf(fmaxf(sc[0], sc[1]), fmaxf(sc[2], sc[3]));
#pragma unroll
    for (int off = 16; off > 0; off >>= 1) mx = fmaxf(mx, __shfl_xor(mx, off));

    float p[4]; float sum = 0.f;
#pragma unroll
    for (int j = 0; j < 4; ++j) {
        float e = __expf(sc[j] - mx);
        if (j * 32 + lane >= cnt) e = 0.f;
        p[j] = e;
        sum += e;
    }
#pragma unroll
    for (int off = 16; off > 0; off >>= 1) sum += __shfl_xor(sum, off);
    float inv = 1.f / sum;

    float o0 = 0.f, o1 = 0.f;
#pragma unroll
    for (int j = 0; j < 4; ++j) {
        int base = j * 32;
        if (base >= cnt) break;           // uniform within wave
        float pj = p[j];
        for (int src = 0; src < 32; ++src) {
            int idx = base + src;
            if (idx >= cnt) break;        // uniform within wave
            float pv = __shfl(pj, src);
            const float* vp = V + ((size_t)(b * T_C + s0 + idx)) * DIM_C + h * HD_C;
            o0 += pv * vp[lane];
            o1 += pv * vp[lane + 32];
        }
    }
    O[rowq + lane]      = (_Float16)(o0 * inv);
    O[rowq + lane + 32] = (_Float16)(o1 * inv);
}

// ---------------------------------------------------------------- residual + layerscale
__global__ __launch_bounds__(256)
void xb_residual(const float* __restrict__ x, const float* __restrict__ t,
                 const float* __restrict__ ls, float* __restrict__ out, int n) {
    int i = blockIdx.x * 256 + threadIdx.x;
    if (i < n) out[i] = x[i] + t[i] * ls[i & (DIM_C - 1)];
}

// ---------------------------------------------------------------- launcher
extern "C" void kernel_launch(void* const* d_in, const int* in_sizes, int n_in,
                              void* d_out, int out_size, void* d_ws, size_t ws_size,
                              hipStream_t stream) {
    (void)in_sizes; (void)n_in; (void)out_size; (void)ws_size;

    const float* x    = (const float*)d_in[0];
    const float* n1w  = (const float*)d_in[1];
    const float* n1b  = (const float*)d_in[2];
    const float* n2w  = (const float*)d_in[3];
    const float* n2b  = (const float*)d_in[4];
    const float* wq   = (const float*)d_in[5];
    const float* wk   = (const float*)d_in[6];
    const float* wv   = (const float*)d_in[7];
    const float* wo   = (const float*)d_in[8];
    const float* w1   = (const float*)d_in[9];
    const float* b1   = (const float*)d_in[10];
    const float* w2   = (const float*)d_in[11];
    const float* b2   = (const float*)d_in[12];
    const float* ls1  = (const float*)d_in[13];
    const float* ls2  = (const float*)d_in[14];

    const size_t MB = 1ull << 20;
    char* ws = (char*)d_ws;
    _Float16* wq16   = (_Float16*)(ws + 0 * MB);     //  2 MB
    _Float16* wk16   = (_Float16*)(ws + 2 * MB);     //  2 MB
    _Float16* wv16   = (_Float16*)(ws + 4 * MB);     //  2 MB
    _Float16* wo16   = (_Float16*)(ws + 6 * MB);     //  2 MB
    _Float16* w116   = (_Float16*)(ws + 8 * MB);     //  8 MB
    _Float16* w216   = (_Float16*)(ws + 16 * MB);    //  8 MB
    _Float16* h16    = (_Float16*)(ws + 24 * MB);    //  8 MB (LN1 then LN2 output)
    float*    Qb     = (float*)   (ws + 32 * MB);    // 16 MB
    float*    Kb     = (float*)   (ws + 48 * MB);    // 16 MB
    float*    Vb     = (float*)   (ws + 64 * MB);    // 16 MB
    _Float16* attn16 = (_Float16*)(ws + 80 * MB);    //  8 MB
    float*    tmp    = (float*)   (ws + 88 * MB);    // 16 MB
    float*    x1     = (float*)   (ws + 104 * MB);   // 16 MB
    _Float16* ffn16  = (_Float16*)(ws + 32 * MB);    // 32 MB, reuses Q/K after attention

    const int NTOK = M_C * DIM_C;                    // 4,194,304

    // weights -> f16
    xb_f32_to_f16<<<(1048576 + 255) / 256, 256, 0, stream>>>(wq, wq16, 1048576);
    xb_f32_to_f16<<<(1048576 + 255) / 256, 256, 0, stream>>>(wk, wk16, 1048576);
    xb_f32_to_f16<<<(1048576 + 255) / 256, 256, 0, stream>>>(wv, wv16, 1048576);
    xb_f32_to_f16<<<(1048576 + 255) / 256, 256, 0, stream>>>(wo, wo16, 1048576);
    xb_f32_to_f16<<<(4194304 + 255) / 256, 256, 0, stream>>>(w1, w116, 4194304);
    xb_f32_to_f16<<<(4194304 + 255) / 256, 256, 0, stream>>>(w2, w216, 4194304);

    // LN1
    xb_layernorm_f16<<<M_C, 256, 0, stream>>>(x, n1w, n1b, h16);

    // QKV projections (M=4096, N=1024, K=1024)
    dim3 gsq(DIM_C / 128, M_C / 128);
    xb_gemm_wmma<0><<<gsq, 256, 0, stream>>>(h16, wq16, nullptr, Qb, M_C, DIM_C, DIM_C);
    xb_gemm_wmma<0><<<gsq, 256, 0, stream>>>(h16, wk16, nullptr, Kb, M_C, DIM_C, DIM_C);
    xb_gemm_wmma<0><<<gsq, 256, 0, stream>>>(h16, wv16, nullptr, Vb, M_C, DIM_C, DIM_C);

    // RoPE on Q,K (B*T*H*32 threads)
    xb_rope<<<(B_C * T_C * HEADS_C * 32) / 256, 256, 0, stream>>>(Qb, Kb);

    // Sliding-window attention: one wave per (b,h,t)
    xb_attn<<<(B_C * HEADS_C * T_C) / 4, 128, 0, stream>>>(Qb, Kb, Vb, attn16);

    // O projection + residual w/ layerscale 1
    xb_gemm_wmma<0><<<gsq, 256, 0, stream>>>(attn16, wo16, nullptr, tmp, M_C, DIM_C, DIM_C);
    xb_residual<<<(NTOK + 255) / 256, 256, 0, stream>>>(x, tmp, ls1, x1, NTOK);

    // LN2
    xb_layernorm_f16<<<M_C, 256, 0, stream>>>(x1, n2w, n2b, h16);

    // FFN1: bias + exact GELU, f16 out (M=4096, N=4096, K=1024)
    dim3 gf1(FFN_C / 128, M_C / 128);
    xb_gemm_wmma<2><<<gf1, 256, 0, stream>>>(h16, w116, b1, ffn16, M_C, FFN_C, DIM_C);

    // FFN2: bias, f32 out (M=4096, N=1024, K=4096)
    xb_gemm_wmma<1><<<gsq, 256, 0, stream>>>(ffn16, w216, b2, tmp, M_C, DIM_C, FFN_C);

    // final residual w/ layerscale 2 -> d_out
    xb_residual<<<(NTOK + 255) / 256, 256, 0, stream>>>(x1, tmp, ls2, (float*)d_out, NTOK);
}
